// VariationalGCNDecoder_54065048322436
// MI455X (gfx1250) — matrix operand order repaired and checked by reading, not compiled, not run on hardware
//
#include <hip/hip_runtime.h>

typedef __attribute__((ext_vector_type(16))) _Float16 v16h;
typedef __attribute__((ext_vector_type(8)))  _Float16 v8h;
typedef __attribute__((ext_vector_type(8)))  float    v8f;

#define NNODES 262144   // ROWS(2048) * N_ATOMS(128)
#define L1DIM  512
#define L2DIM  128

__device__ __forceinline__ float leaky(float v) { return v > 0.f ? v : 0.01f * v; }

__device__ __forceinline__ void atomAddF(float* p, float v) {
  (void)__hip_atomic_fetch_add(p, v, __ATOMIC_RELAXED, __HIP_MEMORY_SCOPE_AGENT);
}

// ---------------- inv_transform: h = leaky(x @ W_inv + b_inv), flattened ----------------
__global__ __launch_bounds__(256) void k_node_h(const float* __restrict__ x,
                                                const float* __restrict__ Winv,
                                                const float* __restrict__ binv,
                                                float* __restrict__ nh) {
  int n = blockIdx.x * 256 + threadIdx.x;      // n = row*128 + atom
  int row = n >> 7, atom = n & 127;
  const float* xr = x + row * 64;
  float acc = binv[atom];
#pragma unroll
  for (int k = 0; k < 64; ++k) acc += xr[k] * Winv[k * 128 + atom];
  nh[n] = leaky(acc);
}

// ---------------- degree / dinv ----------------
__global__ __launch_bounds__(256) void k_deg_init(float* __restrict__ deg) {
  int n = blockIdx.x * 256 + threadIdx.x;
  deg[n] = 1.0f;                               // self-loop
}

__global__ __launch_bounds__(256) void k_deg_edges(const int* __restrict__ ei,
                                                   float* __restrict__ deg, int E) {
  int e = blockIdx.x * 256 + threadIdx.x;
  if (e >= E) return;
  atomAddF(&deg[ei[E + e]], 1.0f);
}

__global__ __launch_bounds__(256) void k_dinv(float* __restrict__ d) {
  int n = blockIdx.x * 256 + threadIdx.x;
  d[n] = rsqrtf(d[n]);                         // deg >= 1 always (self-loops)
}

// ---------------- self-loop init for one aggregation pass ----------------
// aOut[n] = aIn[n] * dinv[n]^2 ; optionally bOut[n] = (bIn ? bIn[n] : 1) * dinv[n]^2
__global__ __launch_bounds__(256) void k_self(const float* __restrict__ dinv,
                                              const float* __restrict__ aIn,
                                              float* __restrict__ aOut,
                                              const float* __restrict__ bIn,
                                              float* __restrict__ bOut) {
  int n = blockIdx.x * 256 + threadIdx.x;
  float dd = dinv[n] * dinv[n];
  aOut[n] = aIn[n] * dd;
  if (bOut) bOut[n] = (bIn ? bIn[n] : 1.0f) * dd;
}

// ---------------- edge scatter pass ----------------
// aOut[dst] += aIn[src]*norm ; optionally bOut[dst] += (bIn ? bIn[src] : 1)*norm
__global__ __launch_bounds__(256) void k_edge_pass(const int* __restrict__ ei, int E,
                                                   const float* __restrict__ dinv,
                                                   const float* __restrict__ aIn,
                                                   float* __restrict__ aOut,
                                                   const float* __restrict__ bIn,
                                                   float* __restrict__ bOut) {
  int e = blockIdx.x * 256 + threadIdx.x;
  if (e >= E) return;
  int s = ei[e], d = ei[E + e];
  float nrm = dinv[s] * dinv[d];
  atomAddF(&aOut[d], aIn[s] * nrm);
  if (bOut) atomAddF(&bOut[d], (bIn ? bIn[s] : 1.0f) * nrm);
}

// ---------------- fold Wc1..Wc3/biases through Wl1 into 4 coefficient vectors ----------------
// cvec[0:512)=alpha (coeff of p), [512:1024)=beta (q), [1024:1536)=gamma (r), [1536:2048)=delta
__global__ __launch_bounds__(256) void k_consts(const float* __restrict__ Wc1, const float* __restrict__ bc1,
                                                const float* __restrict__ Wc2, const float* __restrict__ bc2,
                                                const float* __restrict__ Wc3, const float* __restrict__ bc3,
                                                const float* __restrict__ Wl1, const float* __restrict__ bl1,
                                                float* __restrict__ cvec) {
  int j = blockIdx.x * 256 + threadIdx.x;
  if (j >= 512) return;
  float u2[3], v2[3], a3[3], c3[3], d3[3];
#pragma unroll
  for (int c = 0; c < 3; ++c) {
    float ua = 0.f, va = 0.f;
#pragma unroll
    for (int k = 0; k < 9; ++k) { ua += Wc1[k] * Wc2[k * 3 + c]; va += bc1[k] * Wc2[k * 3 + c]; }
    u2[c] = ua; v2[c] = va;
  }
#pragma unroll
  for (int c = 0; c < 3; ++c) {
    float aa = 0.f, cc = 0.f, dd = 0.f;
#pragma unroll
    for (int k = 0; k < 3; ++k) {
      aa += u2[k] * Wc3[k * 3 + c];
      cc += v2[k] * Wc3[k * 3 + c];
      dd += bc2[k] * Wc3[k * 3 + c];
    }
    a3[c] = aa; c3[c] = cc; d3[c] = dd;
  }
  float al = 0.f, be = 0.f, ga = 0.f, de = bl1[j];
#pragma unroll
  for (int c = 0; c < 3; ++c) {
    float w = Wl1[c * 512 + j];
    al += a3[c] * w; be += c3[c] * w; ga += d3[c] * w; de += bc3[c] * w;
  }
  cvec[j] = al; cvec[512 + j] = be; cvec[1024 + j] = ga; cvec[1536 + j] = de;
}

// ---------------- transpose+convert Wl2 [512,128] f32 -> WT [128,512] f16 ----------------
__global__ __launch_bounds__(256) void k_wt(const float* __restrict__ Wl2, _Float16* __restrict__ WT) {
  int t = blockIdx.x * 256 + threadIdx.x;      // t = n*512 + k
  int n = t >> 9, k = t & 511;
  WT[t] = (_Float16)Wl2[k * 128 + n];
}

// ---------------- fused MLP head: (p,q,r) -> l1(leaky) -> WMMA l2(leaky) -> l3(leaky) ----------------
__global__ __launch_bounds__(256) void k_mlp(const float* __restrict__ P, const float* __restrict__ Q,
                                             const float* __restrict__ R, const float* __restrict__ cvec,
                                             const _Float16* __restrict__ WT, const float* __restrict__ bl2,
                                             const float* __restrict__ Wl3, const float* __restrict__ bl3,
                                             float* __restrict__ out) {
  __shared__ _Float16 A1[16 * L1DIM];          // 16 rows x 512, f16 activations of layer 1
  __shared__ _Float16 A2[16 * L2DIM];          // 16 rows x 128, f16 activations of layer 2
  __shared__ float sP[16], sQ[16], sR[16];
  const int tid = threadIdx.x;
  const int m0 = blockIdx.x * 16;

  if (tid < 16) { sP[tid] = P[m0 + tid]; sQ[tid] = Q[m0 + tid]; sR[tid] = R[m0 + tid]; }
  __syncthreads();

  const float* alpha = cvec;
  const float* beta  = cvec + 512;
  const float* gamma = cvec + 1024;
  const float* delta = cvec + 1536;

  // stage 1: layer-1 activations (gcn3 output folded through Wl1)
  for (int idx = tid; idx < 16 * L1DIM; idx += 256) {
    int row = idx >> 9, j = idx & 511;
    float z = sP[row] * alpha[j] + sQ[row] * beta[j] + sR[row] * gamma[j] + delta[j];
    A1[idx] = (_Float16)leaky(z);
  }
  __syncthreads();

  // stage 2: 16x128 = 8 waves x (16x16 tile), K = 512 via v_wmma_f32_16x16x32_f16
  const int lane = tid & 31;
  const int wv   = tid >> 5;
  const int n0   = wv * 16;
  const int mrow = lane & 15;
  const int khA  = (lane >> 4) * 8;            // A: lanes 0-15 -> K 0..7/16..23; 16-31 -> 8..15/24..31
  const int ncol = n0 + (lane & 15);
  const int khB  = (lane >> 4) * 8;            // B: lanes 0-15 -> K 0..7/16..23; 16-31 -> 8..15/24..31
  const _Float16* arow = &A1[mrow * L1DIM];
  const _Float16* bcol = WT + ncol * L1DIM;

  v8f acc = {};
#pragma unroll
  for (int k0 = 0; k0 < L1DIM; k0 += 32) {
    v8h lo = *(const v8h*)(arow + k0 + khA);
    v8h hi = *(const v8h*)(arow + k0 + 16 + khA);
    v16h a, b;
#pragma unroll
    for (int i = 0; i < 8; ++i) { a[i] = lo[i]; a[8 + i] = hi[i]; }
    v8h blo = *(const v8h*)(bcol + k0 + khB);
    v8h bhi = *(const v8h*)(bcol + k0 + 16 + khB);
#pragma unroll
    for (int i = 0; i < 8; ++i) { b[i] = blo[i]; b[8 + i] = bhi[i]; }
    acc = __builtin_amdgcn_wmma_f32_16x16x32_f16(false, a, false, b, (short)0, acc, false, false);
  }

  float bias = bl2[ncol];
#pragma unroll
  for (int rdx = 0; rdx < 8; ++rdx) {          // D: VGPR r -> M = r + 8*(lane>>4), N = lane&15
    int ml = rdx + 8 * (lane >> 4);
    A2[ml * L2DIM + ncol] = (_Float16)leaky(acc[rdx] + bias);
  }
  __syncthreads();

  // stage 3: [16,128] @ [128,3] + bl3, leaky
  if (tid < 48) {
    int m = tid / 3, c = tid % 3;
    const _Float16* ar = &A2[m * L2DIM];
    float s = bl3[c];
#pragma unroll 16
    for (int k = 0; k < L2DIM; ++k) s += (float)ar[k] * Wl3[k * 3 + c];
    out[(m0 + m) * 3 + c] = leaky(s);
  }
}

// ---------------- edge_index passthrough (second tuple output) ----------------
__global__ __launch_bounds__(256) void k_tail(const int* __restrict__ ei, float* __restrict__ out,
                                              int rem, int convertToFloat) {
  int i = blockIdx.x * 256 + threadIdx.x;
  if (i >= rem) return;
  if (convertToFloat) out[3 * NNODES + i] = (float)ei[i];
  else ((int*)out)[3 * NNODES + i] = ei[i];
}

extern "C" void kernel_launch(void* const* d_in, const int* in_sizes, int n_in,
                              void* d_out, int out_size, void* d_ws, size_t ws_size,
                              hipStream_t stream) {
  const float* x    = (const float*)d_in[0];
  const int*   ei   = (const int*)d_in[1];
  const float* Winv = (const float*)d_in[2];
  const float* binv = (const float*)d_in[3];
  const float* Wc1  = (const float*)d_in[4];
  const float* bc1  = (const float*)d_in[5];
  const float* Wc2  = (const float*)d_in[6];
  const float* bc2  = (const float*)d_in[7];
  const float* Wc3  = (const float*)d_in[8];
  const float* bc3  = (const float*)d_in[9];
  const float* Wl1  = (const float*)d_in[10];
  const float* bl1  = (const float*)d_in[11];
  const float* Wl2  = (const float*)d_in[12];
  const float* bl2  = (const float*)d_in[13];
  const float* Wl3  = (const float*)d_in[14];
  const float* bl3  = (const float*)d_in[15];

  const int E = in_sizes[1] / 2;

  float* ws   = (float*)d_ws;
  float* dinv = ws;                  // holds deg, then dinv in-place
  float* nh   = ws + 1 * NNODES;
  float* S    = ws + 2 * NNODES;
  float* T    = ws + 3 * NNODES;
  float* Rr   = ws + 4 * NNODES;
  float* Pp   = ws + 5 * NNODES;
  float* Qq   = ws + 6 * NNODES;
  float* cvec = ws + 7 * NNODES;                     // 2048 floats
  _Float16* WT = (_Float16*)(ws + 7 * NNODES + 2048); // 65536 halves

  dim3 blk(256);
  const int gN = NNODES / 256;
  const int gE = (E + 255) / 256;

  k_node_h<<<gN, blk, 0, stream>>>(x, Winv, binv, nh);
  k_deg_init<<<gN, blk, 0, stream>>>(dinv);
  k_deg_edges<<<gE, blk, 0, stream>>>(ei, dinv, E);
  k_dinv<<<gN, blk, 0, stream>>>(dinv);

  // pass 1: s = segsum(norm * h[src])   (self loop folded into init)
  k_self<<<gN, blk, 0, stream>>>(dinv, nh, S, nullptr, nullptr);
  k_edge_pass<<<gE, blk, 0, stream>>>(ei, E, dinv, nh, S, nullptr, nullptr);

  // pass 2: t = segsum(norm * s[src]);  r = segsum(norm)
  k_self<<<gN, blk, 0, stream>>>(dinv, S, T, nullptr, Rr);
  k_edge_pass<<<gE, blk, 0, stream>>>(ei, E, dinv, S, T, nullptr, Rr);

  // pass 3: p = segsum(norm * t[src]);  q = segsum(norm * r[src])
  k_self<<<gN, blk, 0, stream>>>(dinv, T, Pp, Rr, Qq);
  k_edge_pass<<<gE, blk, 0, stream>>>(ei, E, dinv, T, Pp, Rr, Qq);

  k_consts<<<2, blk, 0, stream>>>(Wc1, bc1, Wc2, bc2, Wc3, bc3, Wl1, bl1, cvec);
  k_wt<<<(L2DIM * L1DIM) / 256, blk, 0, stream>>>(Wl2, WT);

  k_mlp<<<NNODES / 16, blk, 0, stream>>>(Pp, Qq, Rr, cvec, WT, bl2, Wl3, bl3, (float*)d_out);

  const int rem = out_size - 3 * NNODES;
  if (rem > 0) {
    int conv = (rem == in_sizes[1]) ? 1 : 0;  // int32 values converted; otherwise raw word copy
    k_tail<<<(rem + 255) / 256, blk, 0, stream>>>(ei, (float*)d_out, rem, conv);
  }
}
